// Attention_39865886442202
// MI455X (gfx1250) — matrix operand order repaired and checked by reading, not compiled, not run on hardware
//
#include <hip/hip_runtime.h>

// ---------------- types ----------------
typedef __bf16  v16bf __attribute__((ext_vector_type(16)));
typedef float   v8f   __attribute__((ext_vector_type(8)));

#define DEV __device__ __forceinline__

// ---------------- constants ----------------
constexpr int BB   = 8;      // batch
constexpr int HH   = 8;      // heads
constexpr int NN   = 2048;   // sequence
constexpr int DH   = 32;     // head dim
constexpr int DIM  = 256;
constexpr int QKVC = 768;
constexpr int MTOT = BB * NN;           // 16384 rows
constexpr float SCALE = 0.17677669529663687f;   // 32^-0.5
constexpr float NEGBIG = -3.0e38f;

// workspace element offsets (unsigned short units)
constexpr size_t OFF_XB  = 0;
constexpr size_t SZ_XB   = (size_t)MTOT * DIM;              // 4,194,304
constexpr size_t OFF_WQT = OFF_XB  + SZ_XB;
constexpr size_t SZ_WQT  = (size_t)QKVC * DIM;              // 196,608
constexpr size_t OFF_WPT = OFF_WQT + SZ_WQT;
constexpr size_t SZ_WPT  = (size_t)DIM * DIM;               // 65,536
constexpr size_t OFF_QB  = OFF_WPT + SZ_WPT;
constexpr size_t SZ_HEADS = (size_t)BB * HH * NN * DH;      // 4,194,304
constexpr size_t OFF_KB  = OFF_QB + SZ_HEADS;
constexpr size_t OFF_VB  = OFF_KB + SZ_HEADS;
constexpr size_t OFF_AB  = OFF_VB + SZ_HEADS;

// ---------------- helpers ----------------
DEV unsigned short f2bf(float f) {
    union { float f; unsigned u; } a; a.f = f;
    unsigned u = a.u;
    unsigned r = (u + 0x7FFFu + ((u >> 16) & 1u)) >> 16;   // RNE
    return (unsigned short)r;
}
DEV unsigned pack2(float lo, float hi) {
    return (unsigned)f2bf(lo) | ((unsigned)f2bf(hi) << 16);
}
// 16 bf16 from two contiguous 16-byte chunks
DEV v16bf ldab(const unsigned short* p0, const unsigned short* p1) {
    union { uint4 q[2]; v16bf v; } t;
    t.q[0] = *(const uint4*)p0;
    t.q[1] = *(const uint4*)p1;
    return t.v;
}
DEV v8f wmma_bf16(v16bf a, v16bf b, v8f c) {
    return __builtin_amdgcn_wmma_f32_16x16x32_bf16(
        /*neg_a=*/false, a, /*neg_b=*/false, b,
        /*c_mod=*/(short)0, c, /*reuse_a=*/false, /*reuse_b=*/false);
}

// ---------------- prep kernels ----------------
__global__ void k_cvt_x(const float* __restrict__ x, unsigned short* __restrict__ xb, int n) {
    int i = blockIdx.x * blockDim.x + threadIdx.x;
    if (i < n) xb[i] = f2bf(x[i]);
}
// w[rows][cols] -> wt[cols][rows] (bf16)
__global__ void k_trans(const float* __restrict__ w, unsigned short* __restrict__ wt,
                        int rows, int cols) {
    int i = blockIdx.x * blockDim.x + threadIdx.x;
    if (i < rows * cols) {
        int c = i / rows, k = i % rows;
        wt[i] = f2bf(w[(size_t)k * cols + c]);
    }
}

// ---------------- QKV GEMM: [16384,256] x [256,768] ----------------
// one 32x64 output macro-tile per wave (2 A tiles x 4 B tiles, 8 f32 accumulators)
// writes Q (scaled) / K row-major, V transposed, all bf16
__global__ void k_qkv(const unsigned short* __restrict__ xb,
                      const unsigned short* __restrict__ wt,
                      unsigned short* __restrict__ qb,
                      unsigned short* __restrict__ kb,
                      unsigned short* __restrict__ vb) {
    const int lane = threadIdx.x & 31, wv = threadIdx.x >> 5;
    const int tile = blockIdx.x * 8 + wv;                 // 6144 macro-tiles
    const int m0 = (tile / 12) * 32, c0 = (tile % 12) * 64;
    const int l16 = lane & 15, half = lane >> 4;

    const unsigned short* ap0 = xb + (size_t)(m0 + l16) * DIM + half * 8;
    const unsigned short* ap1 = ap0 + (size_t)16 * DIM;
    const unsigned short* bp  = wt + (size_t)(c0 + l16) * DIM + half * 8;

    v8f acc[8];
#pragma unroll
    for (int t = 0; t < 8; ++t) acc[t] = (v8f){};

#pragma unroll
    for (int k = 0; k < DIM; k += 32) {
        v16bf a0 = ldab(ap0 + k, ap0 + k + 16);
        v16bf a1 = ldab(ap1 + k, ap1 + k + 16);
#pragma unroll
        for (int ci = 0; ci < 4; ++ci) {
            const unsigned short* bpc = bp + (size_t)ci * 16 * DIM;
            v16bf b = ldab(bpc + k, bpc + k + 16);
            acc[ci]     = wmma_bf16(a0, b, acc[ci]);
            acc[4 + ci] = wmma_bf16(a1, b, acc[4 + ci]);
        }
    }
#pragma unroll
    for (int ci = 0; ci < 4; ++ci) {
        const int col = c0 + ci * 16 + l16;
        const int which = col >> 8, rem = col & 255;
        const int h = rem >> 5, d = rem & 31;
#pragma unroll
        for (int mi = 0; mi < 2; ++mi) {
#pragma unroll
            for (int r = 0; r < 8; ++r) {
                int row = m0 + mi * 16 + r + half * 8;
                int b_ = row >> 11, n_ = row & (NN - 1);
                float v = acc[mi * 4 + ci][r];
                size_t hidx = (size_t)(b_ * HH + h);
                if (which == 0)      qb[(hidx * NN + n_) * DH + d] = f2bf(v * SCALE);
                else if (which == 1) kb[(hidx * NN + n_) * DH + d] = f2bf(v);
                else                 vb[(hidx * DH + d) * NN + n_] = f2bf(v);
            }
        }
    }
}

// ---------------- attention: transposed flash w/ periodic band mask ----------------
__global__ void k_attn(const unsigned short* __restrict__ qb,
                       const unsigned short* __restrict__ kb,
                       const unsigned short* __restrict__ vb,
                       unsigned short* __restrict__ ab) {
    const int lane = threadIdx.x & 31, wv = threadIdx.x >> 5;
    const int wid = blockIdx.x * 8 + wv;                  // 8192 waves
    const int qt = wid & 127, h = (wid >> 7) & 7, b = wid >> 10;
    const int q0 = qt * 16;
    const int l16 = lane & 15, half = lane >> 4;

    const size_t headR = (size_t)(b * HH + h) * NN * DH;  // row-major Q/K base
    const size_t headV = (size_t)(b * HH + h) * DH * NN;  // transposed V base

    // Q^T as B operand: lanes 0-15 -> dims 0..7,16..23, lanes 16-31 -> dims 8..15,24..31
    const unsigned short* qp = qb + headR + (size_t)(q0 + l16) * DH + half * 8;
    const v16bf Bq = ldab(qp, qp + 16);

    const int i0 = q0 & 127;
    const int i  = i0 + l16;
    const int lo_i = min(max(i - 5, 0), 117);
    const int jlo  = min(max(i0 - 5, 0), 117);
    const int jhi  = min(max(i0 + 10, 0), 117) + 10;
    const int jc0  = jlo & ~31;

    float m = -1.0e30f, l = 0.0f;
    v8f o0 = {}, o1 = {};

    for (int p = 0; p < 16; ++p) {
        for (int jc = jc0; jc <= jhi; jc += 32) {
            const int k0 = p * 128 + jc;                  // global key base, 32-aligned
            // K tiles as A operand (16 keys x 32 dims)
            const unsigned short* kp = kb + headR + (size_t)(k0 + l16) * DH + half * 8;
            __builtin_prefetch((const void*)(kp + 128 * DH), 0, 1);  // next period
            v16bf Ak0 = ldab(kp, kp + 16);
            v16bf Ak1 = ldab(kp + 16 * DH, kp + 16 * DH + 16);
            v8f s0 = {}, s1 = {};
            s0 = wmma_bf16(Ak0, Bq, s0);                  // S^T keys 0..15 x queries
            s1 = wmma_bf16(Ak1, Bq, s1);                  // S^T keys 16..31
            // mask: allowed iff (j%128) in [lo_i, lo_i+10]
#pragma unroll
            for (int r = 0; r < 8; ++r) {
                int j0 = jc + r + half * 8;
                s0[r] = ((unsigned)(j0 - lo_i) <= 10u) ? s0[r] : NEGBIG;
                int j1 = j0 + 16;
                s1[r] = ((unsigned)(j1 - lo_i) <= 10u) ? s1[r] : NEGBIG;
            }
            // online softmax (per query = lane%16, halves combined via shfl_xor 16)
            float cm = s0[0];
#pragma unroll
            for (int r = 1; r < 8; ++r) cm = fmaxf(cm, s0[r]);
#pragma unroll
            for (int r = 0; r < 8; ++r) cm = fmaxf(cm, s1[r]);
            cm = fmaxf(cm, __shfl_xor(cm, 16, 32));
            float mn = fmaxf(m, cm);
            float alpha = __expf(m - mn);
            unsigned e0[4], e1[4];
            float ssum = 0.0f;
#pragma unroll
            for (int r = 0; r < 4; ++r) {
                float a0 = __expf(s0[2 * r] - mn), a1 = __expf(s0[2 * r + 1] - mn);
                float b0 = __expf(s1[2 * r] - mn), b1 = __expf(s1[2 * r + 1] - mn);
                ssum += a0 + a1 + b0 + b1;
                e0[r] = pack2(a0, a1);
                e1[r] = pack2(b0, b1);
            }
            ssum += __shfl_xor(ssum, 16, 32);
            l = l * alpha + ssum;
            m = mn;
#pragma unroll
            for (int r = 0; r < 8; ++r) { o0[r] *= alpha; o1[r] *= alpha; }
            // build P^T B operand from the lane's own C tiles (keys 8h..8h+7, 16+8h..16+8h+7); swapped values unused
            union { unsigned u[8]; v16bf v; } Pb;
#pragma unroll
            for (int r = 0; r < 4; ++r) {
                unsigned e0s = (unsigned)__shfl_xor((int)e0[r], 16, 32);
                unsigned e1s = (unsigned)__shfl_xor((int)e1[r], 16, 32);
                Pb.u[r]     = e0[r];
                Pb.u[4 + r] = e1[r];
            }
            // V^T tiles as A operand (16 dims x 32 keys), lanes 0-15 keys 0..7,16..23, lanes 16-31 keys 8..15,24..31
            const unsigned short* vp0 = vb + headV + (size_t)l16 * NN + k0 + half * 8;
            v16bf Av0 = ldab(vp0, vp0 + 16);
            const unsigned short* vp1 = vp0 + 16 * NN;
            v16bf Av1 = ldab(vp1, vp1 + 16);
            o0 = wmma_bf16(Av0, Pb.v, o0);                // dims 0..15
            o1 = wmma_bf16(Av1, Pb.v, o1);                // dims 16..31
        }
    }
    const float linv = 1.0f / l;
    // O^T element (r,lane): dim = r + half*8 (+16 for o1), query = lane%16
    const size_t rowbase = ((size_t)b * NN + q0 + l16) * DIM + h * DH;
#pragma unroll
    for (int r = 0; r < 8; ++r) {
        ab[rowbase + r + half * 8]      = f2bf(o0[r] * linv);
        ab[rowbase + 16 + r + half * 8] = f2bf(o1[r] * linv);
    }
}

// ---------------- projection: [16384,256] x [256,256] + bias -> f32 ----------------
// one 32x64 output macro-tile per wave
__global__ void k_proj(const unsigned short* __restrict__ ab,
                       const unsigned short* __restrict__ wt,
                       const float* __restrict__ bias,
                       float* __restrict__ out) {
    const int lane = threadIdx.x & 31, wv = threadIdx.x >> 5;
    const int tile = blockIdx.x * 8 + wv;                 // 2048 macro-tiles
    const int m0 = (tile >> 2) * 32, c0 = (tile & 3) * 64;
    const int l16 = lane & 15, half = lane >> 4;

    const unsigned short* ap0 = ab + (size_t)(m0 + l16) * DIM + half * 8;
    const unsigned short* ap1 = ap0 + (size_t)16 * DIM;
    const unsigned short* bp  = wt + (size_t)(c0 + l16) * DIM + half * 8;

    v8f acc[8];
#pragma unroll
    for (int t = 0; t < 8; ++t) acc[t] = (v8f){};

#pragma unroll
    for (int k = 0; k < DIM; k += 32) {
        v16bf a0 = ldab(ap0 + k, ap0 + k + 16);
        v16bf a1 = ldab(ap1 + k, ap1 + k + 16);
#pragma unroll
        for (int ci = 0; ci < 4; ++ci) {
            const unsigned short* bpc = bp + (size_t)ci * 16 * DIM;
            v16bf b = ldab(bpc + k, bpc + k + 16);
            acc[ci]     = wmma_bf16(a0, b, acc[ci]);
            acc[4 + ci] = wmma_bf16(a1, b, acc[4 + ci]);
        }
    }
#pragma unroll
    for (int ci = 0; ci < 4; ++ci) {
        const int col = c0 + ci * 16 + l16;
        const float bv = bias[col];
#pragma unroll
        for (int mi = 0; mi < 2; ++mi) {
#pragma unroll
            for (int r = 0; r < 8; ++r) {
                int row = m0 + mi * 16 + r + half * 8;
                out[(size_t)row * DIM + col] = acc[mi * 4 + ci][r] + bv;
            }
        }
    }
}

// ---------------- launch ----------------
extern "C" void kernel_launch(void* const* d_in, const int* in_sizes, int n_in,
                              void* d_out, int out_size, void* d_ws, size_t ws_size,
                              hipStream_t stream) {
    const float* x     = (const float*)d_in[0];
    const float* Wqkv  = (const float*)d_in[2];
    const float* Wproj = (const float*)d_in[3];
    const float* bproj = (const float*)d_in[4];
    float* out = (float*)d_out;

    unsigned short* ws = (unsigned short*)d_ws;
    unsigned short* xb  = ws + OFF_XB;
    unsigned short* wqt = ws + OFF_WQT;
    unsigned short* wpt = ws + OFF_WPT;
    unsigned short* qb  = ws + OFF_QB;
    unsigned short* kb  = ws + OFF_KB;
    unsigned short* vb  = ws + OFF_VB;
    unsigned short* ab  = ws + OFF_AB;

    k_cvt_x<<<(int)(SZ_XB / 256), 256, 0, stream>>>(x, xb, (int)SZ_XB);
    k_trans<<<(int)(SZ_WQT / 256), 256, 0, stream>>>(Wqkv, wqt, DIM, QKVC);
    k_trans<<<(int)(SZ_WPT / 256), 256, 0, stream>>>(Wproj, wpt, DIM, DIM);

    k_qkv<<<(MTOT / 32) * (QKVC / 64) / 8, 256, 0, stream>>>(xb, wqt, qb, kb, vb);
    k_attn<<<(BB * HH * (NN / 16)) / 8, 256, 0, stream>>>(qb, kb, vb, ab);
    k_proj<<<(MTOT / 32) * (DIM / 64) / 8, 256, 0, stream>>>(ab, wpt, bproj, out);
}
